// RNN_45028437131357
// MI455X (gfx1250) — matrix operand order repaired and checked by reading, not compiled, not run on hardware
//
#include <hip/hip_runtime.h>
#include <hip/hip_bf16.h>

// ---------------------------------------------------------------------------
// Stacked 2-layer Elman RNN on MI455X (gfx1250).
//   B=2048, T=256, H=512, OUT=1.
// Persistent batch-tiled kernel: 64 WGs x 32 batch rows, time loop in-kernel.
// GEMMs via V_WMMA_F32_16X16X32_F16 (f16 in, f32 acc); weights pre-packed in
// WMMA B-register order, streamed from L2 each step via GLOBAL loads with
// one-K-block software prefetch. Anti-LICM via an opaque zero *offset* so the
// loads keep the global address space (no FLAT path, no scratch spills).
// ---------------------------------------------------------------------------

typedef __attribute__((ext_vector_type(16))) _Float16 v16h;
typedef __attribute__((ext_vector_type(8)))  float    v8f;

#define H_DIM       512
#define T_STEPS     256
#define B_TOT       2048
#define BT          32            // batch rows per workgroup (2 M-tiles)
#define KB_CNT      (H_DIM / 32)  // 16 K-blocks of 32
#define NB_CNT      (H_DIM / 16)  // 32 N-tiles of 16
#define LDS_STRIDE  520           // halves per h row (1040B = 260 dw, %64 = 4)
#define PACKED_MAT_U32 (NB_CNT * KB_CNT * 32 * 8)  // 131072 u32 = 512 KB / matrix

__device__ __forceinline__ v8f wmma_f16(v16h a, v16h b, v8f c) {
  // (neg_a, A, neg_b, B, c_mod, C, reuse_a, reuse_b)
  return __builtin_amdgcn_wmma_f32_16x16x32_f16(false, a, false, b, (short)0, c,
                                                false, false);
}

__device__ __forceinline__ float tanh_fast(float x) {
#if __has_builtin(__builtin_amdgcn_tanhf)
  return __builtin_amdgcn_tanhf(x);          // V_TANH_F32 transcendental
#elif __has_builtin(__builtin_amdgcn_tanh_f32)
  return __builtin_amdgcn_tanh_f32(x);
#else
  // tanh(x) = 1 - 2/(exp(2x)+1): exact identity, v_exp_f32 fast path.
  return 1.0f - 2.0f / (__expf(2.0f * x) + 1.0f);
#endif
}

// A fragment (16x32 f16) from LDS row-major tile.
// ISA A layout: lanes 0-15 hold K=[0..7]|[16..23], lanes 16-31 K=[8..15]|[24..31].
__device__ __forceinline__ v16h ldsA(const _Float16* sh, int row, int kOff, int half) {
  union { uint4 q[2]; v16h v; } u;
  const _Float16* base = sh + row * LDS_STRIDE + kOff + half * 8;
  u.q[0] = *(const uint4*)(base);        // VGPR0-3: 8 consecutive K
  u.q[1] = *(const uint4*)(base + 16);   // VGPR4-7: 8 consecutive K (+16)
  return u.v;
}

// B fragment (32x16 f16) from pre-packed weights: 32B/lane, 1KB/wave, coalesced.
__device__ __forceinline__ v16h gblB(const unsigned* w, int nb, int kb, int lane) {
  const unsigned* p = w + (((nb * KB_CNT) + kb) * 32 + lane) * 8;
  union { uint4 q[2]; v16h v; } u;
  u.q[0] = *(const uint4*)(p);
  u.q[1] = *(const uint4*)(p + 4);
  return u.v;
}

// ---------------------------------------------------------------------------
// Pack one H x H f32 matrix (row-major [K][N]) into f16 B-matrix register
// order: element (K,N) -> u32 index ((nb*KB + kb)*32 + lane)*8 + v, with
//   K = kb*32 + (lane>>4)*8 + (v>>2)*8 + 2v (+1 in high half), N = nb*16 + (lane&15).
// ---------------------------------------------------------------------------
__global__ void pack_weights(const float* __restrict__ Wh0,
                             const float* __restrict__ Wx1,
                             const float* __restrict__ Wh1,
                             unsigned* __restrict__ dst) {
  const int m = blockIdx.y;
  const float* W = (m == 0) ? Wh0 : ((m == 1) ? Wx1 : Wh1);
  unsigned* d = dst + m * PACKED_MAT_U32;

  int idx = blockIdx.x * blockDim.x + threadIdx.x;  // 0 .. 131071
  int v    = idx & 7;
  int lane = (idx >> 3) & 31;
  int kb   = (idx >> 8) & (KB_CNT - 1);
  int nb   = idx >> 12;
  int K = kb * 32 + ((lane >> 4) << 3) + ((v >> 2) << 3) + (v << 1);
  int N = (nb << 4) + (lane & 15);

  union { _Float16 h[2]; unsigned u; } pk;
  pk.h[0] = (_Float16)W[K * H_DIM + N];
  pk.h[1] = (_Float16)W[(K + 1) * H_DIM + N];
  d[idx] = pk.u;
}

// ---------------------------------------------------------------------------
// Persistent RNN kernel: one WG = 32 batch rows, 256 threads (8 waves).
// Wave w: N-columns [w*64, w*64+64), both M-tiles (rows 0-15, 16-31).
// ---------------------------------------------------------------------------
__global__ void __launch_bounds__(256, 2)
rnn_persistent(const float* __restrict__ x,
               const float* __restrict__ Wx0,
               const float* __restrict__ bx0,
               const float* __restrict__ bh0,
               const float* __restrict__ bx1,
               const float* __restrict__ bh1,
               const unsigned* __restrict__ Wh0p,
               const unsigned* __restrict__ Wx1p,
               const unsigned* __restrict__ Wh1p,
               const float* __restrict__ Wfc,
               const float* __restrict__ bfc,
               float* __restrict__ out) {
  __shared__ _Float16 sh0[BT * LDS_STRIDE];   // h0 state, f16 (33.3 KB)
  __shared__ _Float16 sh1[BT * LDS_STRIDE];   // h1 state, f16 (33.3 KB)
  __shared__ float    sb0[H_DIM], sb1[H_DIM], swx0[H_DIM];
  __shared__ float    sxcur[BT];
  __shared__ float    spart[256];

  const int tid   = threadIdx.x;
  const int wave  = tid >> 5;
  const int lane  = tid & 31;
  const int lrow  = lane & 15;   // A row (M, within tile) / C column (N)
  const int lhalf = lane >> 4;
  const int nb0   = wave * 4;    // 4 N-tiles of 16 per wave -> 64 columns
  const int bbase = blockIdx.x * BT;

  // ---- one-time setup ------------------------------------------------------
  for (int i = tid; i < H_DIM; i += 256) {
    sb0[i]  = bx0[i] + bh0[i];
    sb1[i]  = bx1[i] + bh1[i];
    swx0[i] = Wx0[i];
  }
  for (int i = tid; i < BT * LDS_STRIDE; i += 256) {
    sh0[i] = (_Float16)0.0f;
    sh1[i] = (_Float16)0.0f;
  }
  __syncthreads();

  const v8f vzero = {0.f, 0.f, 0.f, 0.f, 0.f, 0.f, 0.f, 0.f};

  // 2 M-tiles x 4 N-tiles accumulators (64 VGPRs); B fragments prefetched one
  // K-block ahead so the 8 WMMAs of block kb overlap the L2 latency of kb+1.
  // 384 v_wmma per wave per timestep.
#define GEMM_ACC(SH, WP)                                                  \
  {                                                                       \
    v16h bf0 = gblB((WP), nb0 + 0, 0, lane);                              \
    v16h bf1 = gblB((WP), nb0 + 1, 0, lane);                              \
    v16h bf2 = gblB((WP), nb0 + 2, 0, lane);                              \
    v16h bf3 = gblB((WP), nb0 + 3, 0, lane);                              \
    _Pragma("unroll")                                                     \
    for (int kb = 0; kb < KB_CNT; ++kb) {                                 \
      v16h nf0 = bf0, nf1 = bf1, nf2 = bf2, nf3 = bf3;                    \
      if (kb + 1 < KB_CNT) {                                              \
        nf0 = gblB((WP), nb0 + 0, kb + 1, lane);                          \
        nf1 = gblB((WP), nb0 + 1, kb + 1, lane);                          \
        nf2 = gblB((WP), nb0 + 2, kb + 1, lane);                          \
        nf3 = gblB((WP), nb0 + 3, kb + 1, lane);                          \
      }                                                                   \
      v16h a0 = ldsA((SH), lrow,      kb * 32, lhalf);                    \
      v16h a1 = ldsA((SH), lrow + 16, kb * 32, lhalf);                    \
      acc[0][0] = wmma_f16(a0, bf0, acc[0][0]);                           \
      acc[1][0] = wmma_f16(a1, bf0, acc[1][0]);                           \
      acc[0][1] = wmma_f16(a0, bf1, acc[0][1]);                           \
      acc[1][1] = wmma_f16(a1, bf1, acc[1][1]);                           \
      acc[0][2] = wmma_f16(a0, bf2, acc[0][2]);                           \
      acc[1][2] = wmma_f16(a1, bf2, acc[1][2]);                           \
      acc[0][3] = wmma_f16(a0, bf3, acc[0][3]);                           \
      acc[1][3] = wmma_f16(a1, bf3, acc[1][3]);                           \
      bf0 = nf0; bf1 = nf1; bf2 = nf2; bf3 = nf3;                         \
    }                                                                     \
  }

#pragma unroll 1
  for (int t = 0; t < T_STEPS; ++t) {
    // Opaque zero OFFSET (not pointer): the weight loads cannot be hoisted
    // out of the time loop (no 1.5MB scratch spill), but the pointers still
    // trace to kernel args, so the loads keep the GLOBAL path / LOADcnt.
    unsigned toff = 0;
    asm volatile("" : "+s"(toff));
    const unsigned* wp_h0 = Wh0p + toff;
    const unsigned* wp_x1 = Wx1p + toff;
    const unsigned* wp_h1 = Wh1p + toff;

    // ---- layer 0: acc = h0 @ Wh0 ----------------------------------------
    v8f acc[2][4] = {{vzero, vzero, vzero, vzero}, {vzero, vzero, vzero, vzero}};
    GEMM_ACC(sh0, wp_h0);

    if (tid < BT) sxcur[tid] = x[(bbase + tid) * T_STEPS + t];
    __syncthreads();  // all h0 reads done; x staged

    // epilogue 0: h0 = tanh(acc + x*Wx0 + bx0 + bh0)
#pragma unroll
    for (int mt = 0; mt < 2; ++mt) {
#pragma unroll
      for (int nt = 0; nt < 4; ++nt) {
        int j = ((nb0 + nt) << 4) + lrow;
        float wx = swx0[j], bb = sb0[j];
#pragma unroll
        for (int p = 0; p < 8; ++p) {
          int r = (mt << 4) + (lhalf << 3) + p;
          float vv = acc[mt][nt][p] + sxcur[r] * wx + bb;
          sh0[r * LDS_STRIDE + j] = (_Float16)tanh_fast(vv);
        }
      }
    }
    __syncthreads();  // new h0 visible

    // ---- layer 1: acc = h0_new @ Wx1 + h1 @ Wh1 --------------------------
#pragma unroll
    for (int mt = 0; mt < 2; ++mt)
#pragma unroll
      for (int nt = 0; nt < 4; ++nt) acc[mt][nt] = vzero;
    GEMM_ACC(sh0, wp_x1);
    GEMM_ACC(sh1, wp_h1);
    __syncthreads();  // all h1 reads done

    // epilogue 1: h1 = tanh(acc + bx1 + bh1)
#pragma unroll
    for (int mt = 0; mt < 2; ++mt) {
#pragma unroll
      for (int nt = 0; nt < 4; ++nt) {
        int j = ((nb0 + nt) << 4) + lrow;
        float bb = sb1[j];
#pragma unroll
        for (int p = 0; p < 8; ++p) {
          int r = (mt << 4) + (lhalf << 3) + p;
          sh1[r * LDS_STRIDE + j] = (_Float16)tanh_fast(acc[mt][nt][p] + bb);
        }
      }
    }
    __syncthreads();  // new h1 visible for next step
  }
#undef GEMM_ACC

  // ---- final: out[b] = h1[b,:] @ Wfc + bfc --------------------------------
  {
    int row  = tid >> 3;   // 0..31
    int part = tid & 7;    // 8 parts x 64 elems
    float s = 0.0f;
#pragma unroll 4
    for (int j = part * 64; j < part * 64 + 64; ++j)
      s += (float)sh1[row * LDS_STRIDE + j] * Wfc[j];
    spart[tid] = s;
    __syncthreads();
    if (tid < BT) {
      float tot = bfc[0];
#pragma unroll
      for (int i = 0; i < 8; ++i) tot += spart[tid * 8 + i];
      out[bbase + tid] = tot;
    }
  }
}

// ---------------------------------------------------------------------------
extern "C" void kernel_launch(void* const* d_in, const int* in_sizes, int n_in,
                              void* d_out, int out_size, void* d_ws, size_t ws_size,
                              hipStream_t stream) {
  (void)in_sizes; (void)n_in; (void)out_size; (void)ws_size;

  const float* x   = (const float*)d_in[0];
  const float* Wx0 = (const float*)d_in[1];
  const float* bx0 = (const float*)d_in[2];
  const float* Wh0 = (const float*)d_in[3];
  const float* bh0 = (const float*)d_in[4];
  const float* Wx1 = (const float*)d_in[5];
  const float* bx1 = (const float*)d_in[6];
  const float* Wh1 = (const float*)d_in[7];
  const float* bh1 = (const float*)d_in[8];
  const float* Wfc = (const float*)d_in[9];
  const float* bfc = (const float*)d_in[10];

  unsigned* wp = (unsigned*)d_ws;  // 3 x 512KB packed f16 weights

  // Pack Wh0 / Wx1 / Wh1 into WMMA B-register order (f16).
  pack_weights<<<dim3(PACKED_MAT_U32 / 256, 3), 256, 0, stream>>>(Wh0, Wx1, Wh1, wp);

  // Persistent time-loop RNN: 64 WGs x 32 batch rows.
  rnn_persistent<<<B_TOT / BT, 256, 0, stream>>>(
      x, Wx0, bx0, bh0, bx1, bh1,
      wp, wp + PACKED_MAT_U32, wp + 2 * PACKED_MAT_U32,
      Wfc, bfc, (float*)d_out);
}
